// EncoderLayer_19061064860195
// MI455X (gfx1250) — matrix operand order repaired and checked by reading, not compiled, not run on hardware
//
#include <hip/hip_runtime.h>
#include <hip/hip_bf16.h>

// ---------------------------------------------------------------------------
// EncoderLayer for MI455X (gfx1250, wave32, WMMA).
// Strategy: f16 WMMA (f32 accumulate) for proj/attn/fc GEMMs, fp32 gates/FFN/softmax/layernorm.
// The u-projection score term is folded into attention via concat:
//   scores_h = [Q_h | uQ*s_q] . [K_h | uK*s_k]^T   (d=128)
// attn output (536MB) is written exactly once -> memory-bound at ~23us floor.
// ---------------------------------------------------------------------------

typedef _Float16 v16h __attribute__((ext_vector_type(16)));
typedef _Float16 v8h  __attribute__((ext_vector_type(8)));
typedef float    v8f  __attribute__((ext_vector_type(8)));

__device__ __forceinline__ v8f wmma16(v16h a, v16h b, v8f c) {
    // (neg_a, A, neg_b, B, c_mod, C, reuse_a, reuse_b)
    return __builtin_amdgcn_wmma_f32_16x16x32_f16(false, a, false, b,
                                                  (short)0, c, false, false);
}

// A operand (16x32 f16): lane holds row m = lane&15; K = {c..c+7, c+16..c+23},
// c = (lane>>4)*8. rowbase points at element (m, kchunk*32 + c).
__device__ __forceinline__ v16h a_from_f16(const _Float16* rowbase) {
    v8h lo = *(const v8h*)rowbase;
    v8h hi = *(const v8h*)(rowbase + 16);
    v16h r;
#pragma unroll
    for (int i = 0; i < 8; ++i) { r[i] = lo[i]; r[8 + i] = hi[i]; }
    return r;
}

__device__ __forceinline__ v16h a_from_f32(const float* rowbase) {
    v16h r;
#pragma unroll
    for (int i = 0; i < 8; ++i) {
        r[i]     = (_Float16)rowbase[i];
        r[8 + i] = (_Float16)rowbase[16 + i];
    }
    return r;
}

// B operand (32x16 f16): lane holds column n = lane&15; K = {koff..koff+7,
// koff+16..koff+23}, koff = (lane>>4)*8.  colrowbase = &Src[row(=n)][kchunk*32+koff]
// for row-major sources where the "row" is the B column index.
__device__ __forceinline__ v16h b_from_f16(const _Float16* colrowbase) {
    return a_from_f16(colrowbase);
}

// Pre-swizzled packed-weight B tile: region + ((kc*NC+nc)*32 + lane)*16 halfs.
__device__ __forceinline__ v16h b_pack(const _Float16* region, int NC,
                                       int kc, int nc, int lane) {
    return *(const v16h*)(region + (((kc * NC + nc) * 32 + lane) << 4));
}

// ---------------------------------------------------------------------------
// Kernel 1: per-batch scalar gates  s = relu(u @ w1) @ w2   (one wave / batch)
// ---------------------------------------------------------------------------
__global__ __launch_bounds__(32) void gates_kernel(
    const float* __restrict__ u,
    const float* __restrict__ sq1, const float* __restrict__ sq2,
    const float* __restrict__ sk1, const float* __restrict__ sk2,
    const float* __restrict__ sv1, const float* __restrict__ sv2,
    float* __restrict__ g) {
    int b = blockIdx.x, j = threadIdx.x;  // j in [0,32)
    const float* ub = u + b * 64;
    float hq = 0.f, hk = 0.f, hv = 0.f;
    for (int d = 0; d < 64; ++d) {
        float x = ub[d];
        hq += x * sq1[d * 32 + j];
        hk += x * sk1[d * 32 + j];
        hv += x * sv1[d * 32 + j];
    }
    float pq = fmaxf(hq, 0.f) * sq2[j];
    float pk = fmaxf(hk, 0.f) * sk2[j];
    float pv = fmaxf(hv, 0.f) * sv2[j];
#pragma unroll
    for (int s = 16; s; s >>= 1) {
        pq += __shfl_xor(pq, s, 32);
        pk += __shfl_xor(pk, s, 32);
        pv += __shfl_xor(pv, s, 32);
    }
    if (j == 0) { g[b] = pq; g[128 + b] = pk; g[256 + b] = pv; }
}

// ---------------------------------------------------------------------------
// Kernel 2: convert a row-major fp32 weight [K,N] into f16 WMMA-B-swizzled
// blocks: tile(kc,nc) -> 32 lanes x 16 halfs; lane: n = nc*16+(ln&15),
// k = kc*32 + (ln>>4)*8 + (s&7) + (s>>3)*16.
// ---------------------------------------------------------------------------
__global__ __launch_bounds__(256) void pack_b16(
    const float* __restrict__ W, int K, int N, _Float16* __restrict__ dst) {
    int idx = blockIdx.x * blockDim.x + threadIdx.x;
    int total = (K >> 5) * (N >> 4) * 512;
    if (idx >= total) return;
    int s = idx & 15, ln = (idx >> 4) & 31, tile = idx >> 9;
    int NC = N >> 4, nc = tile % NC, kc = tile / NC;
    int k = kc * 32 + ((ln >> 4) << 3) + (s & 7) + ((s >> 3) << 4);
    int n = nc * 16 + (ln & 15);
    dst[idx] = (_Float16)W[k * N + n];
}

// ---------------------------------------------------------------------------
// Kernel 3: projections.  256 threads = 8 waves; wave -> 16-token tile.
// Writes f16: Qh/Kh/Vp [B,H,L,64], uQ/uK [B,L,64] (gated).
// ---------------------------------------------------------------------------
__global__ __launch_bounds__(256) void proj_kernel(
    const float* __restrict__ X,
    const _Float16* __restrict__ wqp, const _Float16* __restrict__ wkp,
    const _Float16* __restrict__ wvp, const _Float16* __restrict__ uqp,
    const _Float16* __restrict__ ukp, const _Float16* __restrict__ uvp,
    const float* __restrict__ gates,
    _Float16* __restrict__ Qh, _Float16* __restrict__ Kh,
    _Float16* __restrict__ Vp, _Float16* __restrict__ uQ,
    _Float16* __restrict__ uK) {
    __shared__ __attribute__((aligned(32))) _Float16 swq[16384];
    __shared__ __attribute__((aligned(32))) _Float16 swk[16384];
    __shared__ __attribute__((aligned(32))) _Float16 swv[16384];
    __shared__ __attribute__((aligned(32))) _Float16 suq[4096];
    __shared__ __attribute__((aligned(32))) _Float16 suk[4096];
    __shared__ __attribute__((aligned(32))) _Float16 suv[4096];

    const int tid = threadIdx.x;
    for (int i = tid; i < 2048; i += 256) {  // 16384/8 chunks
        ((v8h*)swq)[i] = ((const v8h*)wqp)[i];
        ((v8h*)swk)[i] = ((const v8h*)wkp)[i];
        ((v8h*)swv)[i] = ((const v8h*)wvp)[i];
    }
    for (int i = tid; i < 512; i += 256) {   // 4096/8 chunks
        ((v8h*)suq)[i] = ((const v8h*)uqp)[i];
        ((v8h*)suk)[i] = ((const v8h*)ukp)[i];
        ((v8h*)suv)[i] = ((const v8h*)uvp)[i];
    }
    __syncthreads();

    const int w = tid >> 5, lane = tid & 31;
    const int t0 = blockIdx.x * 128 + w * 16;     // token tile base
    const int b = t0 >> 9, l0 = t0 & 511;
    const float sq = gates[b], sk = gates[128 + b], sv_ = gates[256 + b];
    const int m = lane & 15, c = (lane >> 4) * 8;
    const int half_ = lane >> 4, col = lane & 15;

    const float* xrow = X + (size_t)(t0 + m) * 64;
    v16h a0 = a_from_f32(xrow + c);
    v16h a1 = a_from_f32(xrow + 32 + c);

    // uV tiles (kept in registers, combined into V' below)
    v8f uv[4];
#pragma unroll
    for (int nc = 0; nc < 4; ++nc) {
        v8f acc = {};
        acc = wmma16(a0, b_pack(suv, 4, 0, nc, lane), acc);
        acc = wmma16(a1, b_pack(suv, 4, 1, nc, lane), acc);
        uv[nc] = acc;
    }

    // gated uQ / uK (stored once per token, shared by all heads)
#pragma unroll
    for (int nc = 0; nc < 4; ++nc) {
        v8f aq = {};
        aq = wmma16(a0, b_pack(suq, 4, 0, nc, lane), aq);
        aq = wmma16(a1, b_pack(suq, 4, 1, nc, lane), aq);
        v8f ak = {};
        ak = wmma16(a0, b_pack(suk, 4, 0, nc, lane), ak);
        ak = wmma16(a1, b_pack(suk, 4, 1, nc, lane), ak);
        _Float16* uqb = uQ + (size_t)t0 * 64 + nc * 16 + col;
        _Float16* ukb = uK + (size_t)t0 * 64 + nc * 16 + col;
#pragma unroll
        for (int r = 0; r < 8; ++r) {
            uqb[(size_t)(r + half_ * 8) * 64] = (_Float16)(aq[r] * sq);
            ukb[(size_t)(r + half_ * 8) * 64] = (_Float16)(ak[r] * sk);
        }
    }

    // per-head Q, K, V' = V + s_v * uV
    for (int h = 0; h < 4; ++h) {
#pragma unroll
        for (int nc = 0; nc < 4; ++nc) {
            int ncg = h * 4 + nc;
            v8f q = {};
            q = wmma16(a0, b_pack(swq, 16, 0, ncg, lane), q);
            q = wmma16(a1, b_pack(swq, 16, 1, ncg, lane), q);
            v8f k = {};
            k = wmma16(a0, b_pack(swk, 16, 0, ncg, lane), k);
            k = wmma16(a1, b_pack(swk, 16, 1, ncg, lane), k);
            v8f v = {};
            v = wmma16(a0, b_pack(swv, 16, 0, ncg, lane), v);
            v = wmma16(a1, b_pack(swv, 16, 1, ncg, lane), v);
            size_t base = ((size_t)(b * 4 + h) * 512 + l0) * 64 + nc * 16 + col;
#pragma unroll
            for (int r = 0; r < 8; ++r) {
                size_t o = base + (size_t)(r + half_ * 8) * 64;
                Qh[o] = (_Float16)q[r];
                Kh[o] = (_Float16)k[r];
                Vp[o] = (_Float16)(v[r] + sv_ * uv[nc][r]);
            }
        }
    }
}

// ---------------------------------------------------------------------------
// Kernel 4: attention.  Grid = B*H*8, block = 128 (4 waves).
// WG handles (b,h) and 64 queries; each wave a 16-query tile with its full
// 16x512 fp32 score row-block in LDS.  V'(b,h) staged once in LDS.
// ---------------------------------------------------------------------------
__global__ __launch_bounds__(128) void attn_kernel(
    const _Float16* __restrict__ Qh, const _Float16* __restrict__ Kh,
    const _Float16* __restrict__ Vp, const _Float16* __restrict__ uQ,
    const _Float16* __restrict__ uK, const int* __restrict__ mask,
    float* __restrict__ attn_out, _Float16* __restrict__ ctx) {
    __shared__ __attribute__((aligned(32))) float sc[4][16][520];  // 133120 B
    __shared__ _Float16 sv[512][66];                               //  67584 B

    const int tid = threadIdx.x;
    const int w = tid >> 5, lane = tid & 31;
    const int qblk = blockIdx.x & 7, bh = blockIdx.x >> 3;
    const int b = bh >> 2, h = bh & 3;
    const int q0 = qblk * 64 + w * 16;

    // stage V'(b,h): 512x64 f16 -> LDS (padded rows)
    const _Float16* vsrc = Vp + (size_t)bh * 512 * 64;
    for (int i = tid; i < 4096; i += 128) {  // 512*64/8 chunks
        int k = i >> 3, n0 = (i & 7) * 8;
        v8h v = *(const v8h*)(vsrc + k * 64 + n0);
#pragma unroll
        for (int j = 0; j < 8; ++j) sv[k][n0 + j] = v[j];
    }
    __syncthreads();

    const int m = lane & 15, c = (lane >> 4) * 8, koff = (lane >> 4) * 8;
    const int half_ = lane >> 4, col = lane & 15;

    // A operands: [Q_h | uQ] for the 16 q rows (d = 128)
    v16h aq[4];
    {
        const _Float16* qrow = Qh + ((size_t)bh * 512 + q0 + m) * 64;
        const _Float16* urow = uQ + ((size_t)b * 512 + q0 + m) * 64;
        aq[0] = a_from_f16(qrow + c);
        aq[1] = a_from_f16(qrow + 32 + c);
        aq[2] = a_from_f16(urow + c);
        aq[3] = a_from_f16(urow + 32 + c);
    }

    // scores: 32 key tiles x 4 WMMA (K=128)
    for (int kb = 0; kb < 32; ++kb) {
        int key = kb * 16 + m;  // B column index == lane&15
        const _Float16* krow  = Kh + ((size_t)bh * 512 + key) * 64;
        const _Float16* ukrow = uK + ((size_t)b * 512 + key) * 64;
        v8f acc = {};
        acc = wmma16(aq[0], b_from_f16(krow + koff), acc);
        acc = wmma16(aq[1], b_from_f16(krow + 32 + koff), acc);
        acc = wmma16(aq[2], b_from_f16(ukrow + koff), acc);
        acc = wmma16(aq[3], b_from_f16(ukrow + 32 + koff), acc);
#pragma unroll
        for (int r = 0; r < 8; ++r)
            sc[w][r + half_ * 8][kb * 16 + col] = acc[r];
    }

    // mask + scale + softmax (wave-local, full 512-wide rows), write attn
    const float scale = 0.125f;  // 1/sqrt(64)
    for (int r = 0; r < 16; ++r) {
        int q = q0 + r;
        const int* mrow = mask + ((size_t)b * 512 + q) * 512;
        float* srow = sc[w][r];
        float mx = -3.4e38f;
#pragma unroll
        for (int i = 0; i < 16; ++i) {
            int idx = lane + i * 32;
            float v = mrow[idx] ? -1e9f : srow[idx] * scale;
            srow[idx] = v;
            mx = fmaxf(mx, v);
        }
#pragma unroll
        for (int s = 16; s; s >>= 1) mx = fmaxf(mx, __shfl_xor(mx, s, 32));
        float sum = 0.f;
#pragma unroll
        for (int i = 0; i < 16; ++i) {
            int idx = lane + i * 32;
            float e = __expf(srow[idx] - mx);
            srow[idx] = e;
            sum += e;
        }
#pragma unroll
        for (int s = 16; s; s >>= 1) sum += __shfl_xor(sum, s, 32);
        float inv = 1.f / sum;
        float* arow = attn_out + ((size_t)bh * 512 + q) * 512;
#pragma unroll
        for (int i = 0; i < 16; ++i) {
            int idx = lane + i * 32;
            float p = srow[idx] * inv;
            srow[idx] = p;
            arow[idx] = p;
        }
    }

    // context: (16x512 P) @ (512x64 V')  -> ctx[b, q, h*64 + n]  (f16)
    v8f acc[4] = {};
    for (int kc = 0; kc < 16; ++kc) {
        const float* prow = &sc[w][m][kc * 32 + c];
        v16h a;
#pragma unroll
        for (int i = 0; i < 8; ++i) {
            a[i]     = (_Float16)prow[i];
            a[8 + i] = (_Float16)prow[16 + i];
        }
#pragma unroll
        for (int nc = 0; nc < 4; ++nc) {
            v16h bm;
#pragma unroll
            for (int s = 0; s < 16; ++s)
                bm[s] = sv[kc * 32 + koff + (s & 7) + ((s >> 3) << 4)][nc * 16 + col];
            acc[nc] = wmma16(a, bm, acc[nc]);
        }
    }
#pragma unroll
    for (int nc = 0; nc < 4; ++nc) {
        _Float16* cb = ctx + ((size_t)b * 512 + q0) * 256 + h * 64 + nc * 16 + col;
#pragma unroll
        for (int r = 0; r < 8; ++r)
            cb[(size_t)(r + half_ * 8) * 256] = (_Float16)acc[nc][r];
    }
}

// ---------------------------------------------------------------------------
// Kernel 5: epilogue.  ctx@fc (WMMA) + residual + LN + FFN + LN (fp32 VALU).
// Grid = 1024, block = 128 (4 waves, 64 tokens per WG).
// ---------------------------------------------------------------------------
__global__ __launch_bounds__(128) void epilogue_kernel(
    const _Float16* __restrict__ ctx, const _Float16* __restrict__ fcp,
    const float* __restrict__ X, const float* __restrict__ ff1,
    const float* __restrict__ ff2, float* __restrict__ out) {
    __shared__ __attribute__((aligned(32))) _Float16 sfc[16384];  // 32KB
    __shared__ float sf1[4096];
    __shared__ float sf2[4096];
    __shared__ float se[64][68];

    const int tid = threadIdx.x;
    for (int i = tid; i < 2048; i += 128)
        ((v8h*)sfc)[i] = ((const v8h*)fcp)[i];
    for (int i = tid; i < 4096; i += 128) { sf1[i] = ff1[i]; sf2[i] = ff2[i]; }
    __syncthreads();

    const int w = tid >> 5, lane = tid & 31;
    const int m = lane & 15, c = (lane >> 4) * 8;
    const int half_ = lane >> 4, col = lane & 15;
    const int t0 = blockIdx.x * 64 + w * 16;

    v8f acc[4] = {};
    const _Float16* crow = ctx + (size_t)(t0 + m) * 256;
    for (int kc = 0; kc < 8; ++kc) {
        v16h a = a_from_f16(crow + kc * 32 + c);
#pragma unroll
        for (int nc = 0; nc < 4; ++nc)
            acc[nc] = wmma16(a, b_pack(sfc, 4, kc, nc, lane), acc[nc]);
    }
#pragma unroll
    for (int nc = 0; nc < 4; ++nc)
#pragma unroll
        for (int r = 0; r < 8; ++r) {
            int row = r + half_ * 8;
            se[w * 16 + row][nc * 16 + col] =
                acc[nc][r] + X[(size_t)(t0 + row) * 64 + nc * 16 + col];
        }
    __syncthreads();

    if (tid < 64) {
        float* e = se[tid];
        float mean = 0.f;
        for (int d = 0; d < 64; ++d) mean += e[d];
        mean *= (1.f / 64.f);
        float var = 0.f;
        for (int d = 0; d < 64; ++d) { float t = e[d] - mean; var += t * t; }
        var *= (1.f / 64.f);
        float inv = rsqrtf(var + 1e-5f);
        for (int d = 0; d < 64; ++d) e[d] = (e[d] - mean) * inv;  // enc

        float hreg[64];
        for (int j = 0; j < 64; ++j) {
            float s = 0.f;
            for (int d = 0; d < 64; ++d) s += e[d] * sf1[d * 64 + j];
            hreg[j] = fmaxf(s, 0.f);
        }
        float o[64];
        float mean2 = 0.f;
        for (int d = 0; d < 64; ++d) {
            float s = e[d];  // residual with enc
            for (int j = 0; j < 64; ++j) s += hreg[j] * sf2[j * 64 + d];
            o[d] = s;
            mean2 += s;
        }
        mean2 *= (1.f / 64.f);
        float var2 = 0.f;
        for (int d = 0; d < 64; ++d) { float t = o[d] - mean2; var2 += t * t; }
        var2 *= (1.f / 64.f);
        float inv2 = rsqrtf(var2 + 1e-5f);
        float* orow = out + (size_t)(blockIdx.x * 64 + tid) * 64;
        for (int d = 0; d < 64; ++d) orow[d] = (o[d] - mean2) * inv2;
    }
}

// ---------------------------------------------------------------------------
extern "C" void kernel_launch(void* const* d_in, const int* in_sizes, int n_in,
                              void* d_out, int out_size, void* d_ws, size_t ws_size,
                              hipStream_t stream) {
    (void)in_sizes; (void)n_in; (void)out_size; (void)ws_size;
    const float* X    = (const float*)d_in[0];
    const int*   mask = (const int*)d_in[1];  // jax bool -> integer per harness rule
    const float* U    = (const float*)d_in[2];
    const float* W_Q  = (const float*)d_in[3];
    const float* W_K  = (const float*)d_in[4];
    const float* W_V  = (const float*)d_in[5];
    const float* U_Q  = (const float*)d_in[6];
    const float* U_K  = (const float*)d_in[7];
    const float* U_V  = (const float*)d_in[8];
    const float* sq1  = (const float*)d_in[9];
    const float* sq2  = (const float*)d_in[10];
    const float* sk1  = (const float*)d_in[11];
    const float* sk2  = (const float*)d_in[12];
    const float* sv1  = (const float*)d_in[13];
    const float* sv2  = (const float*)d_in[14];
    const float* fc   = (const float*)d_in[15];
    const float* ff1  = (const float*)d_in[16];
    const float* ff2  = (const float*)d_in[17];

    float* out      = (float*)d_out;
    float* attn_out = out + (size_t)128 * 512 * 64;  // enc first, then attn

    char*  wsb = (char*)d_ws;
    size_t off = 0;
    auto alloc = [&](size_t bytes) -> void* {
        void* p = wsb + off;
        off = (off + bytes + 255) & ~(size_t)255;
        return p;
    };
    float*     gates = (float*)alloc(384 * sizeof(float));
    _Float16*  wqp   = (_Float16*)alloc(16384 * 2);
    _Float16*  wkp   = (_Float16*)alloc(16384 * 2);
    _Float16*  wvp   = (_Float16*)alloc(16384 * 2);
    _Float16*  uqp   = (_Float16*)alloc(4096 * 2);
    _Float16*  ukp   = (_Float16*)alloc(4096 * 2);
    _Float16*  uvp   = (_Float16*)alloc(4096 * 2);
    _Float16*  fcp   = (_Float16*)alloc(16384 * 2);
    const size_t NBHL = (size_t)128 * 4 * 512 * 64;   // 16,777,216
    const size_t NBL  = (size_t)128 * 512 * 64;       //  4,194,304
    _Float16*  Qh  = (_Float16*)alloc(NBHL * 2);
    _Float16*  Kh  = (_Float16*)alloc(NBHL * 2);
    _Float16*  Vp  = (_Float16*)alloc(NBHL * 2);
    _Float16*  uQg = (_Float16*)alloc(NBL * 2);
    _Float16*  uKg = (_Float16*)alloc(NBL * 2);
    _Float16*  ctx = (_Float16*)alloc((size_t)128 * 512 * 256 * 2);

    gates_kernel<<<128, 32, 0, stream>>>(U, sq1, sq2, sk1, sk2, sv1, sv2, gates);
    pack_b16<<<64, 256, 0, stream>>>(W_Q, 64, 256, wqp);
    pack_b16<<<64, 256, 0, stream>>>(W_K, 64, 256, wkp);
    pack_b16<<<64, 256, 0, stream>>>(W_V, 64, 256, wvp);
    pack_b16<<<16, 256, 0, stream>>>(U_Q, 64, 64, uqp);
    pack_b16<<<16, 256, 0, stream>>>(U_K, 64, 64, ukp);
    pack_b16<<<16, 256, 0, stream>>>(U_V, 64, 64, uvp);
    pack_b16<<<64, 256, 0, stream>>>(fc, 256, 64, fcp);

    proj_kernel<<<512, 256, 0, stream>>>(X, wqp, wkp, wvp, uqp, ukp, uvp,
                                         gates, Qh, Kh, Vp, uQg, uKg);
    attn_kernel<<<4096, 128, 0, stream>>>(Qh, Kh, Vp, uQg, uKg, mask,
                                          attn_out, ctx);
    epilogue_kernel<<<1024, 128, 0, stream>>>(ctx, fcp, X, ff1, ff2, out);
}
